// ScaledDotProductAttention_56135222559414
// MI455X (gfx1250) — matrix operand order repaired and checked by reading, not compiled, not run on hardware
//
#include <hip/hip_runtime.h>
#include <hip/hip_bf16.h>

typedef __attribute__((ext_vector_type(16))) _Float16 v16h;
typedef __attribute__((ext_vector_type(8)))  _Float16 v8h;
typedef __attribute__((ext_vector_type(8)))  float    v8f;
typedef __attribute__((ext_vector_type(4)))  float    v4f;

#define D_EMB 512
#define D_RED 64
#define SEQ   4096
#define BATCH 4

__device__ __forceinline__ v8f wmma16(v16h a, v16h b, v8f c) {
  // D = A(16x32 f16) * B(32x16 f16) + C(16x16 f32)
  return __builtin_amdgcn_wmma_f32_16x16x32_f16(false, a, false, b, (short)0, c,
                                                false, false);
}

// Async 16B global -> LDS copy (per lane), tracked by ASYNCcnt.
__device__ __forceinline__ void async_b128(unsigned lds_off, const void* gaddr) {
  asm volatile("global_load_async_to_lds_b128 %0, %1, off"
               :: "v"(lds_off), "v"((unsigned long long)(size_t)gaddr)
               : "memory");
}

// ---------------------------------------------------------------------------
// Kernel 0: transpose + convert weights: W[512][64] f32 -> Wt[64][512] f16
// ---------------------------------------------------------------------------
__global__ void wtrans_kernel(const float* __restrict__ W,
                              _Float16* __restrict__ Wt) {
  int idx = blockIdx.x * blockDim.x + threadIdx.x;
  if (idx < D_EMB * D_RED) {
    int k = idx >> 6;
    int r = idx & 63;
    Wt[r * D_EMB + k] = (_Float16)W[idx];
  }
}

// ---------------------------------------------------------------------------
// Kernel 1: fused projection GEMM  Y = X[16384,512] @ W[512,64] + bias
// 256 threads = 8 waves; wave handles 16 rows x 64 cols, K-loop over 512.
// X is streamed once -> non-temporal loads keep L2 for W / Q / K / V.
// ---------------------------------------------------------------------------
__global__ void proj_kernel(const float* __restrict__ X,
                            const _Float16* __restrict__ Wt,
                            const float* __restrict__ bias,
                            _Float16* __restrict__ Y,
                            int transpose_out) {
  const int wave  = threadIdx.x >> 5;
  const int lane  = threadIdx.x & 31;
  const int laneM = lane & 15;
  const int hi    = (lane >> 4) & 1;
  const int koffA = hi * 16;
  const int koffB = hi * 16;

  const int row0 = blockIdx.x * 128 + wave * 16;
  const int arow = row0 + laneM;

  v8f acc[4] = {v8f{}, v8f{}, v8f{}, v8f{}};

  for (int kk = 0; kk < D_EMB; kk += 32) {
    // ---- A tile: 16x32 f16 (fp32 -> f16, non-temporal stream) ----
    const v4f* pa0 = (const v4f*)(X + (size_t)arow * D_EMB + kk + koffA);
    const v4f* pa1 = (const v4f*)(X + (size_t)arow * D_EMB + kk + 8 + koffA);
    v4f x0 = __builtin_nontemporal_load(pa0);
    v4f x1 = __builtin_nontemporal_load(pa0 + 1);
    v4f x2 = __builtin_nontemporal_load(pa1);
    v4f x3 = __builtin_nontemporal_load(pa1 + 1);
    v16h a;
#pragma unroll
    for (int j = 0; j < 4; ++j) {
      a[j]      = (_Float16)x0[j];
      a[4 + j]  = (_Float16)x1[j];
      a[8 + j]  = (_Float16)x2[j];
      a[12 + j] = (_Float16)x3[j];
    }
#pragma unroll
    for (int t = 0; t < 4; ++t) {
      const int col = t * 16 + laneM;
      v16h b = *(const v16h*)(Wt + (size_t)col * D_EMB + kk + koffB);
      acc[t] = wmma16(a, b, acc[t]);
    }
  }

#pragma unroll
  for (int t = 0; t < 4; ++t) {
    const int col = t * 16 + laneM;
    const float bv = bias[col];
#pragma unroll
    for (int r = 0; r < 8; ++r) {
      const int m = row0 + hi * 8 + r;            // C/D layout: M = r + 8*hi
      const _Float16 val = (_Float16)(acc[t][r] + bv);
      if (!transpose_out) {
        Y[(size_t)m * D_RED + col] = val;
      } else {
        const int b = m >> 12;
        const int s = m & (SEQ - 1);
        Y[((size_t)b * D_RED + col) * SEQ + s] = val;
      }
    }
  }
}

// ---------------------------------------------------------------------------
// Kernel 2: causal flash attention with async-LDS double-buffered K/V tiles.
// grid (SEQ/64, BATCH), 128 threads = 4 waves; wave owns 16 query rows.
// Per 32-key tile: block stages K (32x64) + V^T (64x32) via
// global_load_async_to_lds_b128 (double-buffered, s_wait_asynccnt), then each
// wave: 4 WMMA (Q@K^T), online softmax, P transpose via per-wave LDS,
// 4 WMMA (P@V).
// ---------------------------------------------------------------------------
__global__ void attn_kernel(const _Float16* __restrict__ Qh,
                            const _Float16* __restrict__ Kh,
                            const _Float16* __restrict__ Vt,
                            float* __restrict__ Out) {
  __shared__ __attribute__((aligned(32))) _Float16 kbuf[2][32 * 64];
  __shared__ __attribute__((aligned(32))) _Float16 vbuf[2][64 * 32];
  __shared__ __attribute__((aligned(32))) _Float16 pbuf[4][16 * 32];

  const int tid   = threadIdx.x;
  const int wave  = tid >> 5;
  const int lane  = tid & 31;
  const int laneM = lane & 15;
  const int hi    = (lane >> 4) & 1;
  const int koffA = hi * 16;
  const int koffB = hi * 16;

  const int b   = blockIdx.y;
  const int q0b = blockIdx.x * 64;          // block's first query row
  const int q0  = q0b + wave * 16;          // wave's first query row

  const _Float16* Q = Qh + (size_t)b * SEQ * D_RED;
  const _Float16* K = Kh + (size_t)b * SEQ * D_RED;
  const _Float16* V = Vt + (size_t)b * D_RED * SEQ;

  // ---- load Q A-tiles once (d_r = 64 -> two 16x32 A operands) ----
  const int qrow = q0 + laneM;
  v16h qa0, qa1;
  {
    v8h l0 = *(const v8h*)(Q + (size_t)qrow * D_RED + 0  + koffA);
    v8h h0 = *(const v8h*)(Q + (size_t)qrow * D_RED + 8 + koffA);
    v8h l1 = *(const v8h*)(Q + (size_t)qrow * D_RED + 32 + koffA);
    v8h h1 = *(const v8h*)(Q + (size_t)qrow * D_RED + 40 + koffA);
#pragma unroll
    for (int j = 0; j < 8; ++j) {
      qa0[j] = l0[j]; qa0[8 + j] = h0[j];
      qa1[j] = l1[j]; qa1[8 + j] = h1[j];
    }
  }

  v8f o[4] = {v8f{}, v8f{}, v8f{}, v8f{}};
  float mrow[8], lrow[8];
#pragma unroll
  for (int r = 0; r < 8; ++r) { mrow[r] = -3.0e38f; lrow[r] = 0.0f; }

  _Float16* pb = pbuf[wave];
  const float scale = 0.125f;                // 1/sqrt(64)
  const int ntiles = q0b / 32 + 2;           // keys 0 .. q0b+63

  // Issue one tile's async copies: K tile 32x64 (256 x 16B), V tile 64x32.
  auto issue_tile = [&](int it) {
    const int bufi = it & 1;
    const int kt   = it * 32;
#pragma unroll
    for (int p = 0; p < 2; ++p) {
      const int c   = tid + p * 128;         // K chunk 0..255
      const int key = c >> 3;                // 8 chunks per 64-dim row
      const int sub = c & 7;
      async_b128((unsigned)(size_t)&kbuf[bufi][key * 64 + sub * 8],
                 K + (size_t)(kt + key) * D_RED + sub * 8);
    }
#pragma unroll
    for (int p = 0; p < 2; ++p) {
      const int c   = tid + p * 128;         // V chunk 0..255
      const int d   = c >> 2;                // 4 chunks per 32-key row
      const int sub = c & 3;
      async_b128((unsigned)(size_t)&vbuf[bufi][d * 32 + sub * 8],
                 V + (size_t)d * SEQ + kt + sub * 8);
    }
  };

  issue_tile(0);

  for (int it = 0; it < ntiles; ++it) {
    const int kt   = it * 32;
    const int bufi = it & 1;

    if (it + 1 < ntiles) {
      issue_tile(it + 1);                       // prefetch next tile
      asm volatile("s_wait_asynccnt 0x4" ::: "memory");  // this wave's tile it done
    } else {
      asm volatile("s_wait_asynccnt 0x0" ::: "memory");
    }
    __syncthreads();                            // tile it visible to all waves

    if (kt <= q0 + 15) {                        // wave has unmasked keys here
      const _Float16* kb = kbuf[bufi];
      const _Float16* vb = vbuf[bufi];

      // ---- scores: two 16x16 tiles over 32 keys ----
      v8f s0 = v8f{}, s1 = v8f{};
      {
        const _Float16* kr0 = kb + (size_t)laneM * D_RED;
        const _Float16* kr1 = kb + (size_t)(16 + laneM) * D_RED;
        v16h b00 = *(const v16h*)(kr0 + 0  + koffB);
        v16h b01 = *(const v16h*)(kr0 + 32 + koffB);
        v16h b10 = *(const v16h*)(kr1 + 0  + koffB);
        v16h b11 = *(const v16h*)(kr1 + 32 + koffB);
        s0 = wmma16(qa0, b00, s0);
        s0 = wmma16(qa1, b01, s0);
        s1 = wmma16(qa0, b10, s1);
        s1 = wmma16(qa1, b11, s1);
      }

      const bool domask = (kt + 31 > q0);

      // ---- online softmax per row (M = r + 8*hi) ----
#pragma unroll
      for (int r = 0; r < 8; ++r) {
        const int qg = q0 + hi * 8 + r;
        float e0 = s0[r] * scale;
        float e1 = s1[r] * scale;
        if (domask) {
          if (kt + laneM      > qg) e0 = -3.0e38f;
          if (kt + 16 + laneM > qg) e1 = -3.0e38f;
        }
        float tm = fmaxf(e0, e1);
#pragma unroll
        for (int off = 1; off < 16; off <<= 1)
          tm = fmaxf(tm, __shfl_xor(tm, off, 32));
        const float mnew = fmaxf(mrow[r], tm);
        const float corr = __expf(mrow[r] - mnew);
        mrow[r] = mnew;
        const float p0 = __expf(e0 - mnew);
        const float p1 = __expf(e1 - mnew);
        float rs = p0 + p1;
#pragma unroll
        for (int off = 1; off < 16; off <<= 1)
          rs += __shfl_xor(rs, off, 32);
        lrow[r] = lrow[r] * corr + rs;
#pragma unroll
        for (int t = 0; t < 4; ++t) o[t][r] *= corr;
        const int M = r + hi * 8;
        pb[M * 32 + laneM]      = (_Float16)p0;
        pb[M * 32 + 16 + laneM] = (_Float16)p1;
      }

      // per-wave LDS in-order guarantee; wait for P data
      asm volatile("s_wait_dscnt 0x0" ::: "memory");

      // ---- reload P as 16x32 A-operand ----
      v16h pa;
      {
        v8h lo = *(const v8h*)(pb + laneM * 32 + koffA);
        v8h hh = *(const v8h*)(pb + laneM * 32 + 8 + koffA);
#pragma unroll
        for (int j = 0; j < 8; ++j) { pa[j] = lo[j]; pa[8 + j] = hh[j]; }
      }

      // ---- O += P @ V (V^T tile in LDS, B-layout contiguous) ----
#pragma unroll
      for (int t = 0; t < 4; ++t) {
        const int dcol = t * 16 + laneM;
        v16h bv = *(const v16h*)(vb + (size_t)dcol * 32 + koffB);
        o[t] = wmma16(pa, bv, o[t]);
      }
    }

    __syncthreads();   // everyone done reading buf before it is re-filled
  }

  // ---- finalize: O / l, non-temporal store fp32 ----
#pragma unroll
  for (int t = 0; t < 4; ++t) {
#pragma unroll
    for (int r = 0; r < 8; ++r) {
      const int m = q0 + hi * 8 + r;
      float* p = Out + ((size_t)b * SEQ + m) * D_RED + t * 16 + laneM;
      __builtin_nontemporal_store(o[t][r] / lrow[r], p);
    }
  }
}

// ---------------------------------------------------------------------------
extern "C" void kernel_launch(void* const* d_in, const int* in_sizes, int n_in,
                              void* d_out, int out_size, void* d_ws, size_t ws_size,
                              hipStream_t stream) {
  const float* q_in = (const float*)d_in[0];
  const float* k_in = (const float*)d_in[1];
  const float* v_in = (const float*)d_in[2];
  const float* Wq   = (const float*)d_in[3];
  const float* bq   = (const float*)d_in[4];
  const float* Wk   = (const float*)d_in[5];
  const float* bk   = (const float*)d_in[6];
  const float* Wv   = (const float*)d_in[7];
  const float* bv   = (const float*)d_in[8];
  float* out = (float*)d_out;

  const size_t nQKV = (size_t)BATCH * SEQ * D_RED;
  const size_t nW   = (size_t)D_RED * D_EMB;
  _Float16* Qh  = (_Float16*)d_ws;
  _Float16* Kh  = Qh + nQKV;
  _Float16* Vt  = Kh + nQKV;
  _Float16* Wtq = Vt + nQKV;
  _Float16* Wtk = Wtq + nW;
  _Float16* Wtv = Wtk + nW;

  const int wtThreads = 256;
  const int wtBlocks  = (D_EMB * D_RED + wtThreads - 1) / wtThreads;
  wtrans_kernel<<<wtBlocks, wtThreads, 0, stream>>>(Wq, Wtq);
  wtrans_kernel<<<wtBlocks, wtThreads, 0, stream>>>(Wk, Wtk);
  wtrans_kernel<<<wtBlocks, wtThreads, 0, stream>>>(Wv, Wtv);

  const int projBlocks = (BATCH * SEQ) / 128;
  proj_kernel<<<projBlocks, 256, 0, stream>>>(q_in, Wtq, bq, Qh, 0);
  proj_kernel<<<projBlocks, 256, 0, stream>>>(k_in, Wtk, bk, Kh, 0);
  proj_kernel<<<projBlocks, 256, 0, stream>>>(v_in, Wtv, bv, Vt, 1);

  attn_kernel<<<dim3(SEQ / 64, BATCH), 128, 0, stream>>>(Qh, Kh, Vt, out);
}
